// TransformerDecoderStage_39470749450983
// MI455X (gfx1250) — matrix operand order repaired and checked by reading, not compiled, not run on hardware
//
#include <hip/hip_runtime.h>

#define N_   4
#define L_   1024
#define H_   16
#define DH_  64
#define D_   1024
#define DFF_ 4096

typedef __attribute__((ext_vector_type(16))) __bf16 v16bf;
typedef __attribute__((ext_vector_type(8)))  float  v8f;
typedef __attribute__((ext_vector_type(4)))  int    v4i;

#if defined(__has_builtin)
#  if __has_builtin(__builtin_amdgcn_global_load_async_to_lds_b128)
#    define ATHENA_ASYNC_LDS 1
#  endif
#endif
#ifndef ATHENA_ASYNC_LDS
#  define ATHENA_ASYNC_LDS 0
#endif

union Frag {
  uint4 u4[2];
  unsigned short h[16];
  v16bf v;
};

__device__ __forceinline__ unsigned short f32_to_bf16(float f) {
  unsigned int u = __float_as_uint(f);
  unsigned int r = (u + 0x7FFFu + ((u >> 16) & 1u)) >> 16;
  return (unsigned short)r;
}

__device__ __forceinline__ v8f wmma_bf16(const Frag& a, const Frag& b, v8f c) {
  // D = A*B + C, f32 accumulate, bf16 operands, 16x16x32
  return __builtin_amdgcn_wmma_f32_16x16x32_bf16(false, a.v, false, b.v,
                                                 (short)0, c, false, false);
}

// ---------------------------------------------------------------------------
// fp32 -> bf16 conversion
// ---------------------------------------------------------------------------
__global__ __launch_bounds__(256) void cvt_bf16_kernel(
    const float* __restrict__ in, unsigned short* __restrict__ out, int n) {
  int i = blockIdx.x * 256 + threadIdx.x;
  if (i < n) out[i] = f32_to_bf16(in[i]);
}

// ---------------------------------------------------------------------------
// C[m,n] = sum_k A[m,k] * B[n,k]  (Linear: x @ W.T), optional bias/ReLU.
// A: MxK bf16 row-major, B: NxK bf16 row-major.
// Block: 256 threads = 8 waves; tile 128x128, BK=32; wave tile 64x32.
// OUTMODE: 0 = f32 row-major, 1 = bf16 row-major,
//          2 = bf16 transposed V layout (N_,H_,DH_,L_) for flash attention.
// ---------------------------------------------------------------------------
template <int RELU, int OUTMODE>
__global__ __launch_bounds__(256) void gemm_nt_kernel(
    const unsigned short* __restrict__ A, const unsigned short* __restrict__ B,
    const float* __restrict__ bias, float* Cf, unsigned short* Cb,
    int M, int Nn, int K) {
  __shared__ __align__(16) unsigned short As[128 * 32];
  __shared__ __align__(16) unsigned short Bs[128 * 32];

  const int tid  = threadIdx.x;
  const int lane = tid & 31;
  const int wave = tid >> 5;
  const int wm   = wave >> 2;  // 0..1  (64 rows each)
  const int wn   = wave & 3;   // 0..3  (32 cols each)
  const int lh   = lane >> 4;  // lane half
  const int lr   = lane & 15;
  const int rowBase = blockIdx.y * 128;
  const int colBase = blockIdx.x * 128;
  const int khA = lh * 8;   // A fragment K interleave
  const int khB = lh * 8;   // B fragment K interleave

  v8f acc[4][2];
  for (int m = 0; m < 4; ++m)
    for (int n2 = 0; n2 < 2; ++n2)
      for (int v = 0; v < 8; ++v) acc[m][n2][v] = 0.f;

  // per-thread staging coordinates (each thread moves 2x16B of A and of B)
  const int r0 = tid >> 2;               // 0..63
  const int r1 = r0 + 64;                // 64..127
  const int kc = (tid & 3) * 8;          // 0,8,16,24

  for (int k0 = 0; k0 < K; k0 += 32) {
    const unsigned short* a0 = A + (size_t)(rowBase + r0) * K + k0 + kc;
    const unsigned short* a1 = A + (size_t)(rowBase + r1) * K + k0 + kc;
    const unsigned short* b0 = B + (size_t)(colBase + r0) * K + k0 + kc;
    const unsigned short* b1 = B + (size_t)(colBase + r1) * K + k0 + kc;
    if (k0 + 32 < K) {  // prefetch next K tile into cache hierarchy
      __builtin_prefetch(a0 + 32, 0, 3);
      __builtin_prefetch(b0 + 32, 0, 3);
    }
#if ATHENA_ASYNC_LDS
    // CDNA5 async copy: global -> LDS without VGPR round trip (ASYNCcnt)
    __builtin_amdgcn_global_load_async_to_lds_b128(
        (__attribute__((address_space(1))) v4i*)a0,
        (__attribute__((address_space(3))) v4i*)(&As[r0 * 32 + kc]), 0, 0);
    __builtin_amdgcn_global_load_async_to_lds_b128(
        (__attribute__((address_space(1))) v4i*)a1,
        (__attribute__((address_space(3))) v4i*)(&As[r1 * 32 + kc]), 0, 0);
    __builtin_amdgcn_global_load_async_to_lds_b128(
        (__attribute__((address_space(1))) v4i*)b0,
        (__attribute__((address_space(3))) v4i*)(&Bs[r0 * 32 + kc]), 0, 0);
    __builtin_amdgcn_global_load_async_to_lds_b128(
        (__attribute__((address_space(1))) v4i*)b1,
        (__attribute__((address_space(3))) v4i*)(&Bs[r1 * 32 + kc]), 0, 0);
    asm volatile("s_wait_asynccnt 0x0" ::: "memory");
#else
    *(uint4*)(&As[r0 * 32 + kc]) = *(const uint4*)a0;
    *(uint4*)(&As[r1 * 32 + kc]) = *(const uint4*)a1;
    *(uint4*)(&Bs[r0 * 32 + kc]) = *(const uint4*)b0;
    *(uint4*)(&Bs[r1 * 32 + kc]) = *(const uint4*)b1;
#endif
    __syncthreads();

    Frag af[4], bfr[2];
#pragma unroll
    for (int m = 0; m < 4; ++m) {
      int base = (wm * 64 + m * 16 + lr) * 32 + khA;
      af[m].u4[0] = *(const uint4*)(&As[base]);
      af[m].u4[1] = *(const uint4*)(&As[base + 16]);
    }
#pragma unroll
    for (int n2 = 0; n2 < 2; ++n2) {
      int base = (wn * 32 + n2 * 16 + lr) * 32 + khB;
      bfr[n2].u4[0] = *(const uint4*)(&Bs[base]);
      bfr[n2].u4[1] = *(const uint4*)(&Bs[base + 16]);
    }
#pragma unroll
    for (int m = 0; m < 4; ++m)
#pragma unroll
      for (int n2 = 0; n2 < 2; ++n2)
        acc[m][n2] = wmma_bf16(af[m], bfr[n2], acc[m][n2]);
    __syncthreads();
  }

#pragma unroll
  for (int m = 0; m < 4; ++m)
#pragma unroll
    for (int n2 = 0; n2 < 2; ++n2)
#pragma unroll
      for (int v = 0; v < 8; ++v) {
        int row = rowBase + wm * 64 + m * 16 + v + 8 * lh;
        int col = colBase + wn * 32 + n2 * 16 + lr;
        float x = acc[m][n2][v];
        if (bias) x += bias[col];
        if (RELU) x = fmaxf(x, 0.f);
        if (OUTMODE == 0) {
          Cf[(size_t)row * Nn + col] = x;
        } else if (OUTMODE == 1) {
          Cb[(size_t)row * Nn + col] = f32_to_bf16(x);
        } else {
          // V output: (N_,H_,DH_,L_) so attention reads key-contiguous vectors
          int nb = row >> 10, l = row & (L_ - 1);
          int hh = col >> 6, dh = col & (DH_ - 1);
          size_t idx = ((((size_t)nb * H_ + hh) * DH_) + dh) * L_ + l;
          Cb[idx] = f32_to_bf16(x);
        }
      }
}

// ---------------------------------------------------------------------------
// Flash attention: one wave per (qtile of 16, head, batch). Online softmax.
// Q,K: (N,L,D) bf16 (head h at cols [h*64,h*64+64)); Vt: (N,H,DH,L) bf16.
// scaling = rsqrt(#non-padded keys); causal + key-padding masks.
// ---------------------------------------------------------------------------
template <int CAUSAL>
__global__ __launch_bounds__(32) void flash_attn_kernel(
    const unsigned short* __restrict__ Q, const unsigned short* __restrict__ Km,
    const unsigned short* __restrict__ Vt, const unsigned char* __restrict__ pad,
    float* __restrict__ Out) {
  __shared__ __align__(16) unsigned short Pt[16 * 32];  // P tile (A-layout staging)

  const int lane = threadIdx.x;
  const int qt = blockIdx.x, h = blockIdx.y, n = blockIdx.z;
  const int lh = lane >> 4, lr = lane & 15;
  const int khA = lh * 8, khB = lh * 8;

  // number of valid (non-padded) keys for this sample
  int cnt = 0;
  for (int i = lane; i < L_; i += 32) cnt += (pad[n * L_ + i] == 0) ? 1 : 0;
#pragma unroll
  for (int off = 16; off > 0; off >>= 1) cnt += __shfl_xor(cnt, off, 32);
  const float invs = rsqrtf((float)cnt);

  // Q fragments (A-layout): rows qt*16 + lr, dh split into 2 K-steps of 32
  Frag qf[2];
  {
    const unsigned short* qr =
        Q + ((size_t)(n * L_ + qt * 16 + lr)) * D_ + h * DH_;
    qf[0].u4[0] = *(const uint4*)(qr + khA);
    qf[0].u4[1] = *(const uint4*)(qr + khA + 16);
    qf[1].u4[0] = *(const uint4*)(qr + 32 + khA);
    qf[1].u4[1] = *(const uint4*)(qr + 32 + khA + 16);
  }
  const unsigned short* vbase = Vt + ((size_t)n * H_ + h) * DH_ * L_;

  v8f o[4];
  for (int t = 0; t < 4; ++t)
    for (int v = 0; v < 8; ++v) o[t][v] = 0.f;
  float mrow[8], lrow[8];
  for (int v = 0; v < 8; ++v) { mrow[v] = -__builtin_inff(); lrow[v] = 0.f; }

  int kbEnd = (cnt + 31) >> 5;
  if (CAUSAL) { int c = (qt >> 1) + 1; if (c < kbEnd) kbEnd = c; }

  for (int kb = 0; kb < kbEnd; ++kb) {
    // S = Q x K^T for 32 keys (two 16-col WMMA tiles, K-dim = dh = 2x32)
    v8f s[2];
#pragma unroll
    for (int st = 0; st < 2; ++st) {
      const unsigned short* kr =
          Km + ((size_t)(n * L_ + kb * 32 + st * 16 + lr)) * D_ + h * DH_;
      Frag k0f, k1f;
      k0f.u4[0] = *(const uint4*)(kr + khB);
      k0f.u4[1] = *(const uint4*)(kr + khB + 16);
      k1f.u4[0] = *(const uint4*)(kr + 32 + khB);
      k1f.u4[1] = *(const uint4*)(kr + 32 + khB + 16);
      v8f c;
      for (int v = 0; v < 8; ++v) c[v] = 0.f;
      c = wmma_bf16(qf[0], k0f, c);
      c = wmma_bf16(qf[1], k1f, c);
      s[st] = c;
    }

    // scale + masks + per-row block max (rows live at v + 8*lh; keys at lr)
    float bm[8];
    for (int v = 0; v < 8; ++v) bm[v] = -__builtin_inff();
#pragma unroll
    for (int st = 0; st < 2; ++st)
      for (int v = 0; v < 8; ++v) {
        int qrow = qt * 16 + v + 8 * lh;
        int key  = kb * 32 + st * 16 + lr;
        float x = s[st][v] * invs;
        if (key >= cnt || (CAUSAL && key > qrow)) x = -__builtin_inff();
        s[st][v] = x;
        if (x > bm[v]) bm[v] = x;
      }
#pragma unroll
    for (int off = 8; off > 0; off >>= 1)
      for (int v = 0; v < 8; ++v)
        bm[v] = fmaxf(bm[v], __shfl_xor(bm[v], off, 32));

    // online softmax update
    float sf[8], bs[8];
    for (int v = 0; v < 8; ++v) {
      float nm = fmaxf(mrow[v], bm[v]);
      sf[v] = (mrow[v] == -__builtin_inff()) ? 0.f : __expf(mrow[v] - nm);
      mrow[v] = nm;
      bs[v] = 0.f;
    }
#pragma unroll
    for (int st = 0; st < 2; ++st)
      for (int v = 0; v < 8; ++v) {
        float x = s[st][v];
        float p = (x == -__builtin_inff()) ? 0.f : __expf(x - mrow[v]);
        bs[v] += p;
        Pt[(v + 8 * lh) * 32 + st * 16 + lr] = f32_to_bf16(p);
      }
#pragma unroll
    for (int off = 8; off > 0; off >>= 1)
      for (int v = 0; v < 8; ++v) bs[v] += __shfl_xor(bs[v], off, 32);
    for (int v = 0; v < 8; ++v) lrow[v] = lrow[v] * sf[v] + bs[v];
    for (int t = 0; t < 4; ++t)
      for (int v = 0; v < 8; ++v) o[t][v] *= sf[v];
    __syncthreads();

    // P as A-fragment; V as B-fragments straight from transposed global V
    Frag pf;
    pf.u4[0] = *(const uint4*)(&Pt[lr * 32 + khA]);
    pf.u4[1] = *(const uint4*)(&Pt[lr * 32 + khA + 16]);
#pragma unroll
    for (int t = 0; t < 4; ++t) {
      const unsigned short* vr =
          vbase + (size_t)(t * 16 + lr) * L_ + kb * 32 + khB;
      Frag vf;
      vf.u4[0] = *(const uint4*)(vr);
      vf.u4[1] = *(const uint4*)(vr + 16);
      o[t] = wmma_bf16(pf, vf, o[t]);
    }
    __syncthreads();
  }

#pragma unroll
  for (int t = 0; t < 4; ++t)
#pragma unroll
    for (int v = 0; v < 8; ++v) {
      int row = qt * 16 + v + 8 * lh;
      float l = lrow[v];
      float val = (l > 0.f) ? o[t][v] / l : 0.f;
      Out[((size_t)(n * L_ + row)) * D_ + h * DH_ + t * 16 + lr] = val;
    }
}

// ---------------------------------------------------------------------------
// out = LayerNorm(a + b) * g + beta ; writes f32 (+ optional bf16 copy)
// One 256-thread block per row of D=1024.
// ---------------------------------------------------------------------------
__global__ __launch_bounds__(256) void add_ln_kernel(
    const float* __restrict__ a, const float* b, const float* __restrict__ g,
    const float* __restrict__ beta, float* outf, unsigned short* outb) {
  __shared__ float r1[256];
  __shared__ float r2[256];
  const int row = blockIdx.x;
  const int tid = threadIdx.x;
  const size_t base = (size_t)row * D_;
  float x[4];
  float s = 0.f, s2 = 0.f;
#pragma unroll
  for (int i = 0; i < 4; ++i) {
    int c = i * 256 + tid;
    float v = a[base + c] + b[base + c];
    x[i] = v; s += v; s2 += v * v;
  }
  r1[tid] = s; r2[tid] = s2;
  __syncthreads();
  for (int st = 128; st > 0; st >>= 1) {
    if (tid < st) { r1[tid] += r1[tid + st]; r2[tid] += r2[tid + st]; }
    __syncthreads();
  }
  float mu  = r1[0] * (1.f / D_);
  float var = r2[0] * (1.f / D_) - mu * mu;
  float rstd = rsqrtf(var + 1e-5f);
#pragma unroll
  for (int i = 0; i < 4; ++i) {
    int c = i * 256 + tid;
    float y = (x[i] - mu) * rstd * g[c] + beta[c];
    outf[base + c] = y;
    if (outb) outb[base + c] = f32_to_bf16(y);
  }
}

// ---------------------------------------------------------------------------
extern "C" void kernel_launch(void* const* d_in, const int* in_sizes, int n_in,
                              void* d_out, int out_size, void* d_ws,
                              size_t ws_size, hipStream_t stream) {
  (void)in_sizes; (void)n_in; (void)out_size; (void)ws_size;
  const float* encoded = (const float*)d_in[0];
  const float* Yin     = (const float*)d_in[1];
  // d_in[2] = causal mask (regenerated analytically)
  const unsigned char* pad = (const unsigned char*)d_in[3];
  const float* Wq1 = (const float*)d_in[4];
  const float* Wk1 = (const float*)d_in[5];
  const float* Wv1 = (const float*)d_in[6];
  const float* g1  = (const float*)d_in[7];
  const float* b1  = (const float*)d_in[8];
  const float* Wq2 = (const float*)d_in[9];
  const float* Wk2 = (const float*)d_in[10];
  const float* Wv2 = (const float*)d_in[11];
  const float* g2  = (const float*)d_in[12];
  const float* b2  = (const float*)d_in[13];
  const float* We  = (const float*)d_in[14];
  const float* be  = (const float*)d_in[15];
  const float* Wc  = (const float*)d_in[16];
  const float* bc  = (const float*)d_in[17];
  const float* g3  = (const float*)d_in[18];
  const float* b3  = (const float*)d_in[19];

  char* ws = (char*)d_ws;
  size_t off = 0;
  auto alloc = [&](size_t bytes) -> void* {
    void* p = ws + off;
    off += (bytes + 255) & ~(size_t)255;
    return p;
  };
  const size_t MLD = (size_t)N_ * L_ * D_;  // 4M elements
  unsigned short* Yb   = (unsigned short*)alloc(MLD * 2);
  unsigned short* Eb   = (unsigned short*)alloc(MLD * 2);
  unsigned short* Qb   = (unsigned short*)alloc(MLD * 2);
  unsigned short* Kb   = (unsigned short*)alloc(MLD * 2);
  unsigned short* Vb   = (unsigned short*)alloc(MLD * 2);  // transposed layout
  unsigned short* Wq1b = (unsigned short*)alloc((size_t)D_ * D_ * 2);
  unsigned short* Wk1b = (unsigned short*)alloc((size_t)D_ * D_ * 2);
  unsigned short* Wv1b = (unsigned short*)alloc((size_t)D_ * D_ * 2);
  unsigned short* Wq2b = (unsigned short*)alloc((size_t)D_ * D_ * 2);
  unsigned short* Wk2b = (unsigned short*)alloc((size_t)D_ * D_ * 2);
  unsigned short* Wv2b = (unsigned short*)alloc((size_t)D_ * D_ * 2);
  unsigned short* Web  = (unsigned short*)alloc((size_t)DFF_ * D_ * 2);
  unsigned short* Wcb  = (unsigned short*)alloc((size_t)DFF_ * D_ * 2);
  unsigned short* Hb   = (unsigned short*)alloc((size_t)N_ * L_ * DFF_ * 2);
  float* tmpf = (float*)alloc(MLD * 4);
  float* Ycur = (float*)alloc(MLD * 4);

  const int M = N_ * L_;  // 4096
  dim3 blk256(256), blk32(32);
  auto cvt = [&](const float* src, unsigned short* dst, size_t n) {
    cvt_bf16_kernel<<<dim3((unsigned)((n + 255) / 256)), blk256, 0, stream>>>(
        src, dst, (int)n);
  };
  cvt(Yin, Yb, MLD);
  cvt(encoded, Eb, MLD);
  cvt(Wq1, Wq1b, (size_t)D_ * D_);
  cvt(Wk1, Wk1b, (size_t)D_ * D_);
  cvt(Wv1, Wv1b, (size_t)D_ * D_);
  cvt(Wq2, Wq2b, (size_t)D_ * D_);
  cvt(Wk2, Wk2b, (size_t)D_ * D_);
  cvt(Wv2, Wv2b, (size_t)D_ * D_);
  cvt(We, Web, (size_t)DFF_ * D_);
  cvt(Wc, Wcb, (size_t)DFF_ * D_);

  dim3 gP(D_ / 128, M / 128);  // projection GEMMs (8, 32)
  gemm_nt_kernel<0, 1><<<gP, blk256, 0, stream>>>(Yb, Wq1b, nullptr, nullptr, Qb, M, D_, D_);
  gemm_nt_kernel<0, 1><<<gP, blk256, 0, stream>>>(Yb, Wk1b, nullptr, nullptr, Kb, M, D_, D_);
  gemm_nt_kernel<0, 2><<<gP, blk256, 0, stream>>>(Yb, Wv1b, nullptr, nullptr, Vb, M, D_, D_);

  dim3 gA(L_ / 16, H_, N_);
  flash_attn_kernel<1><<<gA, blk32, 0, stream>>>(Qb, Kb, Vb, pad, tmpf);
  add_ln_kernel<<<dim3(M), blk256, 0, stream>>>(tmpf, Yin, g1, b1, Ycur, Yb);

  gemm_nt_kernel<0, 1><<<gP, blk256, 0, stream>>>(Yb, Wq2b, nullptr, nullptr, Qb, M, D_, D_);
  gemm_nt_kernel<0, 1><<<gP, blk256, 0, stream>>>(Eb, Wk2b, nullptr, nullptr, Kb, M, D_, D_);
  gemm_nt_kernel<0, 2><<<gP, blk256, 0, stream>>>(Eb, Wv2b, nullptr, nullptr, Vb, M, D_, D_);

  flash_attn_kernel<0><<<gA, blk32, 0, stream>>>(Qb, Kb, Vb, pad, tmpf);
  add_ln_kernel<<<dim3(M), blk256, 0, stream>>>(tmpf, Ycur, g2, b2, Ycur, Yb);

  dim3 gF1(DFF_ / 128, M / 128);  // (32, 32)
  gemm_nt_kernel<1, 1><<<gF1, blk256, 0, stream>>>(Yb, Web, be, nullptr, Hb, M, DFF_, D_);
  dim3 gF2(D_ / 128, M / 128);    // (8, 32)
  gemm_nt_kernel<0, 0><<<gF2, blk256, 0, stream>>>(Hb, Wcb, bc, tmpf, nullptr, M, D_, DFF_);

  add_ln_kernel<<<dim3(M), blk256, 0, stream>>>(tmpf, Ycur, g3, b3, (float*)d_out, nullptr);
}
